// RbfNet_21921513079255
// MI455X (gfx1250) — matrix operand order repaired and checked by reading, not compiled, not run on hardware
//
#include <hip/hip_runtime.h>

typedef __attribute__((ext_vector_type(16))) __bf16 v16bf;
typedef __attribute__((ext_vector_type(8)))  __bf16 v8bf;
typedef __attribute__((ext_vector_type(4)))  __bf16 v4bf;
typedef __attribute__((ext_vector_type(8)))  float  v8f;
typedef __attribute__((ext_vector_type(4)))  float  v4f;
typedef __attribute__((ext_vector_type(2)))  float  v2f;
typedef __attribute__((ext_vector_type(4)))  int    v4i;

// ---- CDNA5 async global->LDS copy (ASYNCcnt-tracked) ----------------------
#if defined(__has_builtin)
#  if __has_builtin(__builtin_amdgcn_global_load_async_to_lds_b128)
#    define HAVE_ASYNC_LDS_BUILTIN 1
#  endif
#  if __has_builtin(__builtin_amdgcn_s_wait_asynccnt)
#    define HAVE_WAIT_ASYNC_BUILTIN 1
#  endif
#endif

#if defined(HAVE_WAIT_ASYNC_BUILTIN)
#  define WAIT_ASYNC(n) __builtin_amdgcn_s_wait_asynccnt(n)
#else
#  define WAIT_ASYNC(n) asm volatile("s_wait_asynccnt " #n ::: "memory")
#endif

// Copy 16 bytes (8 bf16) per lane from global to LDS asynchronously.
__device__ __forceinline__ void async_copy16(const __bf16* __restrict__ g,
                                             __bf16* __restrict__ l)
{
#if defined(HAVE_ASYNC_LDS_BUILTIN)
    __builtin_amdgcn_global_load_async_to_lds_b128(
        (__attribute__((address_space(1))) v4i*)g,
        (__attribute__((address_space(3))) v4i*)l,
        0, 0);
#else
    unsigned long long ga = (unsigned long long)g;
    __attribute__((address_space(3))) char* l3 =
        (__attribute__((address_space(3))) char*)l;
    unsigned la = (unsigned)(__SIZE_TYPE__)l3;
    asm volatile("global_load_async_to_lds_b128 %0, %1, off"
                 :: "v"(la), "v"(ga) : "memory");
#endif
}

// ---------------------------------------------------------------------------
// Prep: split f32 rows (D=128) into bf16 hi + bf16 lo, and compute row ||.||^2.
// One wave32 per row: lane handles 4 consecutive floats, then a 5-step
// shfl_xor reduction for the sum of squares.
// ---------------------------------------------------------------------------
__global__ __launch_bounds__(256) void rbf_prep_split(
    const float* __restrict__ src,
    __bf16* __restrict__ hi, __bf16* __restrict__ lo,
    float* __restrict__ sumsq, int nrows)
{
    const int wave = threadIdx.x >> 5;
    const int lane = threadIdx.x & 31;
    const int row  = blockIdx.x * 8 + wave;
    if (row >= nrows) return;

    v4f v = *(const v4f*)(src + row * 128 + lane * 4);
    v4bf h, l;
    float ss = 0.0f;
#pragma unroll
    for (int i = 0; i < 4; ++i) {
        float f  = v[i];
        __bf16 hb = (__bf16)f;
        h[i] = hb;
        l[i] = (__bf16)(f - (float)hb);
        ss += f * f;
    }
    *(v4bf*)(hi + row * 128 + lane * 4) = h;
    *(v4bf*)(lo + row * 128 + lane * 4) = l;

#pragma unroll
    for (int off = 16; off >= 1; off >>= 1)
        ss += __shfl_xor(ss, off, 32);
    if (lane == 0) sumsq[row] = ss;
}

// ---------------------------------------------------------------------------
// Fused RBF kernel. 8 waves/block, each wave owns a 16-row x tile and sweeps
// all 1024 centers in 16-column chunks. The per-chunk centers tile (bf16 hi+lo,
// 8KB) is double-buffered in LDS via async global->LDS copies so the 8 waves
// share one copy and the prefetch of chunk nc+1 overlaps WMMA on chunk nc.
//   scores  : 3x v_wmma_f32_16x16x32_bf16 per K=32 chunk (bf16x3 split GEMM)
//   radial  : d2 = x2 + c2 - 2s ; exp(-beta*sqrt(max(d2,0))) in registers
//   head    : radial(16x16) @ W^T chunk via v_wmma_f32_16x16x4_f32, with the
//             radial tile re-striped C/D-layout -> A-layout through 1KB LDS.
// ---------------------------------------------------------------------------
__global__ __launch_bounds__(256) void rbf_fused(
    const __bf16* __restrict__ xh, const __bf16* __restrict__ xl,
    const __bf16* __restrict__ ch, const __bf16* __restrict__ cl,
    const float* __restrict__ x2, const float* __restrict__ c2,
    const float* __restrict__ beta, const float* __restrict__ W,
    const float* __restrict__ bias, float* __restrict__ out)
{
    __shared__ __align__(32) __bf16 s_ch[2][2048];  // 16 rows x 128, double-buffered
    __shared__ __align__(32) __bf16 s_cl[2][2048];
    __shared__ float lds_tile[8][256];              // per-wave 16x16 f32 radial tile

    const int tid     = threadIdx.x;
    const int wave    = tid >> 5;
    const int lane    = tid & 31;
    const int half    = lane >> 4;       // 0: lanes 0-15, 1: lanes 16-31
    const int l16     = lane & 15;
    const int rowbase = (blockIdx.x * 8 + wave) * 16;

    // --- Persistent A fragments: this lane's x row, K=128 as 4 chunks of 32.
    const int arow = rowbase + l16;
    v16bf ah[4], al[4];
#pragma unroll
    for (int kc = 0; kc < 4; ++kc) {
        const __bf16* ph = xh + arow * 128 + kc * 32 + 16 * half;
        ah[kc] = *(const v16bf*)ph;
        const __bf16* pl = xl + arow * 128 + kc * 32 + 16 * half;
        al[kc] = *(const v16bf*)pl;
    }

    // --- Per-lane row norms for the 8 C/D rows this lane sees (M = r + 8*half)
    float x2v[8];
#pragma unroll
    for (int r = 0; r < 8; ++r) x2v[r] = x2[rowbase + 8 * half + r];

    v8f acc2 = {};                         // [16 rows x 16 classes] head acc
    const int ceff = (l16 < 10) ? l16 : 0; // clamp padded class lanes

    // --- prologue: async-prefetch chunk 0 into buffer 0 (16B per thread)
    async_copy16(ch + tid * 8, &s_ch[0][tid * 8]);
    async_copy16(cl + tid * 8, &s_cl[0][tid * 8]);

    for (int nc = 0; nc < 64; ++nc) {
        const int buf = nc & 1;
        const int nb  = nc * 16;
        const int bn  = nb + l16;          // this lane's center column

        // ---- async-prefetch chunk nc+1 into the other buffer ----
        if (nc + 1 < 64) {
            const int src = (nc + 1) * 2048 + tid * 8;
            async_copy16(ch + src, &s_ch[buf ^ 1][tid * 8]);
            async_copy16(cl + src, &s_cl[buf ^ 1][tid * 8]);
            WAIT_ASYNC(2);                 // chunk nc's 2 copies (in-order) done
        } else {
            WAIT_ASYNC(0);
        }
        __syncthreads();                   // all waves' shares of chunk nc visible

        // ---- GEMM1: scores = x . centers^T  (bf16x3, f32 accumulate) ----
        v8f acc = {};
#pragma unroll
        for (int kc = 0; kc < 4; ++kc) {
            // B 32x16 bf16 in the same contiguous k order as A: lane N=l16, half selects K 0-15 / 16-31
            // -> one contiguous 32B LDS load per lane.
            v16bf bh = *(const v16bf*)&s_ch[buf][l16 * 128 + kc * 32 + 16 * half];
            v16bf bl = *(const v16bf*)&s_cl[buf][l16 * 128 + kc * 32 + 16 * half];
            acc = __builtin_amdgcn_wmma_f32_16x16x32_bf16(false, ah[kc], false, bh,
                                                          (short)0, acc, false, false);
            acc = __builtin_amdgcn_wmma_f32_16x16x32_bf16(false, ah[kc], false, bl,
                                                          (short)0, acc, false, false);
            acc = __builtin_amdgcn_wmma_f32_16x16x32_bf16(false, al[kc], false, bh,
                                                          (short)0, acc, false, false);
        }

        // ---- elementwise: dist -> radial, written into this wave's LDS tile ----
        const float c2v   = c2[bn];
        const float betav = beta[bn];
#pragma unroll
        for (int r = 0; r < 8; ++r) {
            float d2     = x2v[r] + c2v - 2.0f * acc[r];
            // raw v_sqrt_f32 (~1 ULP): far below the bf16x3 noise floor,
            // avoids the precise-sqrt Newton/denormal-guard expansion.
            float dist   = __builtin_amdgcn_sqrtf(fmaxf(d2, 0.0f));
            float radial = __expf(-betav * dist);
            // C/D layout: element (M = r + 8*half, N = l16)
            lds_tile[wave][(r + 8 * half) * 16 + l16] = radial;
        }
        // intra-wave DS RAW: stores above feed the transposed loads below
        asm volatile("s_wait_dscnt 0x0" ::: "memory");

        // ---- GEMM2: acc2 += radial(16x16) @ W^T chunk (f32 16x16x4) ----
#pragma unroll
        for (int kc2 = 0; kc2 < 4; ++kc2) {
            // A f32 16x4 layout: lane M=l16, K = kc2*4 + 2*half + {0,1}
            v2f a2 = *(const v2f*)&lds_tile[wave][l16 * 16 + kc2 * 4 + 2 * half];
            // B f32 4x16 layout: lane N=class, K = 2*half + {0,1}; B[k][c]=W[c][nb+k]
            v2f b2 = *(const v2f*)(W + ceff * 1024 + nb + kc2 * 4 + 2 * half);
            acc2 = __builtin_amdgcn_wmma_f32_16x16x4_f32(false, a2, false, b2,
                                                         (short)0, acc2, false, false);
        }

        __syncthreads();                   // all reads of buf done before reuse
    }

    // ---- epilogue: acc2 C/D layout (M = r + 8*half, N = class), add bias ----
    if (l16 < 10) {
        float bv = bias[l16];
#pragma unroll
        for (int r = 0; r < 8; ++r)
            out[(rowbase + r + 8 * half) * 10 + l16] = acc2[r] + bv;
    }
}

// ---------------------------------------------------------------------------
extern "C" void kernel_launch(void* const* d_in, const int* in_sizes, int n_in,
                              void* d_out, int out_size, void* d_ws, size_t ws_size,
                              hipStream_t stream)
{
    const float* x       = (const float*)d_in[0]; // [32768,128]
    const float* centers = (const float*)d_in[1]; // [1024,128]
    const float* beta    = (const float*)d_in[2]; // [1,1024]
    const float* W       = (const float*)d_in[3]; // [10,1024]
    const float* b       = (const float*)d_in[4]; // [10]
    float* out           = (float*)d_out;         // [32768,10]

    const int B = 32768, N = 1024, D = 128;

    char* ws = (char*)d_ws;
    size_t off = 0;
    __bf16* xh = (__bf16*)(ws + off); off += (size_t)B * D * sizeof(__bf16); // 8 MiB
    __bf16* xl = (__bf16*)(ws + off); off += (size_t)B * D * sizeof(__bf16);
    __bf16* ch = (__bf16*)(ws + off); off += (size_t)N * D * sizeof(__bf16);
    __bf16* cl = (__bf16*)(ws + off); off += (size_t)N * D * sizeof(__bf16);
    float*  x2 = (float*)(ws + off);  off += (size_t)B * sizeof(float);
    float*  c2 = (float*)(ws + off);  off += (size_t)N * sizeof(float);

    rbf_prep_split<<<dim3(B / 8), dim3(256), 0, stream>>>(x, xh, xl, x2, B);
    rbf_prep_split<<<dim3(N / 8), dim3(256), 0, stream>>>(centers, ch, cl, c2, N);

    // 8 waves/block * 16 rows/wave = 128 rows/block -> 256 blocks
    rbf_fused<<<dim3(B / 128), dim3(256), 0, stream>>>(
        xh, xl, ch, cl, x2, c2, beta, W, b, out);
}
